// MyViT_57320633532993
// MI455X (gfx1250) — matrix operand order repaired and checked by reading, not compiled, not run on hardware
//
#include <hip/hip_runtime.h>
#include <hip/hip_bf16.h>
#include <math.h>

// ---------------- model constants ----------------
#define kB   32
#define kC   3
#define kP   14
#define kPS  16
#define kD   768
#define kNH  12
#define kHD  64
#define kL   12
#define kDFF 3072
#define kOUT 1000
#define kS   197           // tokens
#define kNP  196           // patches per image
#define kM   (kB * kS)     // 6304 rows in token GEMMs (= 197*32)
#define kMP  (kB * kNP)    // 6272 rows in patch-embed GEMM (= 196*32)

typedef __bf16 bf16;
typedef __bf16 v16bf __attribute__((ext_vector_type(16)));
typedef __bf16 v8bf  __attribute__((ext_vector_type(8)));
typedef float  v8f   __attribute__((ext_vector_type(8)));

// ---------------- weight convert + transpose: dst[n*K+k] = (bf16)src[k*N+n] ----------------
__global__ __launch_bounds__(256) void transpose_bf16_kernel(
    const float* __restrict__ src, bf16* __restrict__ dst, int K, int N) {
  int idx = blockIdx.x * 256 + threadIdx.x;
  if (idx >= K * N) return;
  int n = idx / K;
  int k = idx - n * K;                       // consecutive threads -> coalesced writes
  dst[idx] = (bf16)src[(size_t)k * N + n];
}

// ---------------- patchify: x[B,C,H,W] -> bf16 patches[B*196, 768] ----------------
__global__ __launch_bounds__(256) void patchify_kernel(
    const float* __restrict__ x, bf16* __restrict__ patches) {
  int idx = blockIdx.x * 256 + threadIdx.x;
  const int total = kMP * (kC * kPS * kPS);
  if (idx >= total) return;
  int row = idx / 768;
  int col = idx - row * 768;
  int b  = row / kNP;
  int p  = row - b * kNP;
  int py = p / kP, px = p - py * kP;
  int c  = col / (kPS * kPS);
  int r  = col - c * (kPS * kPS);
  int iy = r / kPS, ix = r - iy * kPS;
  size_t src = (((size_t)b * kC + c) * 224 + (py * kPS + iy)) * 224 + (px * kPS + ix);
  patches[idx] = (bf16)x[src];
}

// ---------------- assemble: tmp[b,s,d] = (s==0 ? cls[d] : tokens[b,s-1,d]) + pos[s,d] ----------------
__global__ __launch_bounds__(256) void assemble_kernel(
    const float* __restrict__ tokens, const float* __restrict__ cls,
    const float* __restrict__ pos, float* __restrict__ tmp) {
  int idx = blockIdx.x * 256 + threadIdx.x;
  const int total = kM * kD;
  if (idx >= total) return;
  int d   = idx % kD;
  int row = idx / kD;
  int b = row / kS;
  int s = row - b * kS;
  float v = (s == 0) ? cls[d] : tokens[((size_t)b * kNP + (s - 1)) * kD + d];
  tmp[idx] = v + pos[(size_t)s * kD + d];
}

// ---------------- layernorm over D=768; OUT_BF16 selects output buffer ----------------
template <int OUT_BF16>
__global__ __launch_bounds__(256) void layernorm_kernel(
    const float* __restrict__ x, const float* __restrict__ g, const float* __restrict__ bta,
    float* __restrict__ yf, bf16* __restrict__ yb) {
  int row = blockIdx.x, t = threadIdx.x;
  const float* xr = x + (size_t)row * kD;
  float s = 0.f, s2 = 0.f;
  for (int d = t; d < kD; d += 256) { float v = xr[d]; s += v; s2 += v * v; }
  __shared__ float r1[256], r2[256];
  r1[t] = s; r2[t] = s2; __syncthreads();
  for (int o = 128; o > 0; o >>= 1) {
    if (t < o) { r1[t] += r1[t + o]; r2[t] += r2[t + o]; }
    __syncthreads();
  }
  float mean = r1[0] * (1.0f / kD);
  float var  = r2[0] * (1.0f / kD) - mean * mean;
  float rstd = rsqrtf(var + 1e-5f);
  for (int d = t; d < kD; d += 256) {
    float v = (xr[d] - mean) * rstd * g[d] + bta[d];
    if (OUT_BF16) yb[(size_t)row * kD + d] = (bf16)v;
    else          yf[(size_t)row * kD + d] = v;
  }
}

// ---------------- A-fragment load (16-bit A 16x32 lane layout) ----------------
__device__ __forceinline__ v16bf load_afrag(const bf16* __restrict__ p) {
  v8bf lo = *(const v8bf*)p;          // K = base .. base+7   (this lane's half)
  v8bf hi = *(const v8bf*)(p + 8);    // K = base+8 .. +15
  return __builtin_shufflevector(lo, hi,
      0, 1, 2, 3, 4, 5, 6, 7, 8, 9, 10, 11, 12, 13, 14, 15);
}

#define WMMA_BF16(a, b, c) \
  __builtin_amdgcn_wmma_f32_16x16x32_bf16(false, (a), false, (b), (short)0, (c), false, false)

// ---------------- bf16 WMMA GEMM, 32(M)x64(N) register tile per wave ----------------
// C[M,N] = A[M,K] @ Wt[N,K]^T (+bias, epilogue)
// EPI 0: Cf = acc + bias            (fp32 out)
// EPI 1: Cf += acc + bias           (residual accumulate, in-place fp32)
// EPI 2: Cb = bf16(gelu(acc+bias))  (exact erf GELU, bf16 out)
template <int EPI>
__global__ __launch_bounds__(256) void gemm_bf16_kernel(
    const bf16* __restrict__ A, const bf16* __restrict__ Bt,
    const float* __restrict__ bias,
    float* __restrict__ Cf, bf16* __restrict__ Cb,
    int M, int N, int K) {
  const int lane = threadIdx.x & 31;
  const int wave = blockIdx.x * 8 + (threadIdx.x >> 5);
  const int tilesN = N >> 6;                 // 64-wide N tiles
  const int tilesM = M >> 5;                 // 32-high M tiles
  if (wave >= tilesM * tilesN) return;       // whole-wave exit: EXEC stays all-ones
  const int tm = wave / tilesN;
  const int tn = wave - tm * tilesN;

  const int r0 = tm * 32 + (lane & 15);
  const int c0 = tn * 64 + (lane & 15);
  const int ka = (lane >> 4) << 4;           // 0 or 16
  const int kb = (lane >> 4) << 4;           // 0 or 16
  const bf16* ap0 = A  + (size_t)r0 * K + ka;
  const bf16* ap1 = ap0 + (size_t)16 * K;
  const bf16* bp  = Bt + (size_t)c0 * K + kb;
  const size_t bstep = (size_t)16 * K;       // next 16-column group of Wt

  v8f acc0 = {}, acc1 = {}, acc2 = {}, acc3 = {};
  v8f acc4 = {}, acc5 = {}, acc6 = {}, acc7 = {};

#pragma unroll 2
  for (int k = 0; k < K; k += 32) {
    v16bf a0 = load_afrag(ap0 + k);
    v16bf a1 = load_afrag(ap1 + k);
    v16bf b0 = *(const v16bf*)(bp + k);
    v16bf b1 = *(const v16bf*)(bp + bstep + k);
    v16bf b2 = *(const v16bf*)(bp + 2 * bstep + k);
    v16bf b3 = *(const v16bf*)(bp + 3 * bstep + k);
    acc0 = WMMA_BF16(a0, b0, acc0);
    acc1 = WMMA_BF16(a0, b1, acc1);
    acc2 = WMMA_BF16(a0, b2, acc2);
    acc3 = WMMA_BF16(a0, b3, acc3);
    acc4 = WMMA_BF16(a1, b0, acc4);
    acc5 = WMMA_BF16(a1, b1, acc5);
    acc6 = WMMA_BF16(a1, b2, acc6);
    acc7 = WMMA_BF16(a1, b3, acc7);
  }

  // epilogue: C tile rows = tm*32 + i*16 + (lane>=16?8:0) + j, col = tn*64 + jt*16 + lane%16
  const int rb = tm * 32 + ((lane >> 4) << 3);
  const int nb = tn * 64 + (lane & 15);
  v8f accs[8] = {acc0, acc1, acc2, acc3, acc4, acc5, acc6, acc7};
#pragma unroll
  for (int i = 0; i < 2; ++i) {
#pragma unroll
    for (int jt = 0; jt < 4; ++jt) {
      v8f a = accs[i * 4 + jt];
      const int n = nb + jt * 16;
      const float bv = bias[n];
#pragma unroll
      for (int j = 0; j < 8; ++j) {
        float v = a[j] + bv;
        size_t off = (size_t)(rb + i * 16 + j) * N + n;
        if constexpr (EPI == 0) {
          Cf[off] = v;
        } else if constexpr (EPI == 1) {
          Cf[off] += v;
        } else {
          float g = 0.5f * v * (1.0f + erff(v * 0.70710678118654752f));
          Cb[off] = (bf16)g;
        }
      }
    }
  }
}

// ---------------- attention: one 64-thread block per (b, head, query s) ----------------
__global__ __launch_bounds__(64) void attention_kernel(
    const float* __restrict__ qkv, bf16* __restrict__ ctx) {
  const int blk = blockIdx.x;                // ((b*NH)+h)*S + s
  const int s  = blk % kS;
  const int bh = blk / kS;
  const int h  = bh % kNH;
  const int b  = bh / kNH;
  const int t  = threadIdx.x;                // 0..63 (= HD)

  __shared__ float qs[kHD];
  __shared__ float sc[kS];
  __shared__ float red[64];

  const size_t base = (size_t)b * kS * (3 * kD);
  const float* qrow = qkv + base + (size_t)s * (3 * kD) + h * kHD;
  qs[t] = qrow[t];
  __syncthreads();

  const float scale = 0.125f;                // 1/sqrt(64)
  for (int j = t; j < kS; j += 64) {
    const float* krow = qkv + base + (size_t)j * (3 * kD) + kD + h * kHD;
    float a = 0.f;
#pragma unroll 8
    for (int d = 0; d < kHD; ++d) a += qs[d] * krow[d];
    sc[j] = a * scale;
  }
  __syncthreads();

  float m = -1e30f;
  for (int j = t; j < kS; j += 64) m = fmaxf(m, sc[j]);
  red[t] = m; __syncthreads();
  for (int o = 32; o > 0; o >>= 1) { if (t < o) red[t] = fmaxf(red[t], red[t + o]); __syncthreads(); }
  m = red[0];
  __syncthreads();

  float lsum = 0.f;
  for (int j = t; j < kS; j += 64) { float e = expf(sc[j] - m); sc[j] = e; lsum += e; }
  red[t] = lsum; __syncthreads();
  for (int o = 32; o > 0; o >>= 1) { if (t < o) red[t] += red[t + o]; __syncthreads(); }
  const float inv = 1.0f / red[0];
  __syncthreads();

  float a = 0.f;                             // thread t owns output dim d = t
  for (int j = 0; j < kS; ++j) {
    const float* vrow = qkv + base + (size_t)j * (3 * kD) + 2 * kD + h * kHD;
    a += sc[j] * vrow[t];
  }
  ctx[((size_t)(b * kS + s)) * kD + h * kHD + t] = (bf16)(a * inv);
}

// ---------------- head: post-LN on cls token, tanh MLP, 1000 logits; one block per batch ----------------
__global__ __launch_bounds__(256) void head_kernel(
    const float* __restrict__ hbuf,
    const float* __restrict__ pg, const float* __restrict__ pb,
    const float* __restrict__ w1, const float* __restrict__ b1,
    const float* __restrict__ w2, const float* __restrict__ b2,
    float* __restrict__ out) {
  const int b = blockIdx.x, t = threadIdx.x;
  __shared__ float c[kD];
  __shared__ float tt[kD];
  __shared__ float r1[256], r2[256];

  const float* xr = hbuf + (size_t)(b * kS) * kD;  // token 0 (cls)
  float s = 0.f, s2 = 0.f;
  for (int d = t; d < kD; d += 256) { float v = xr[d]; s += v; s2 += v * v; }
  r1[t] = s; r2[t] = s2; __syncthreads();
  for (int o = 128; o > 0; o >>= 1) {
    if (t < o) { r1[t] += r1[t + o]; r2[t] += r2[t + o]; }
    __syncthreads();
  }
  float mean = r1[0] * (1.0f / kD);
  float var  = r2[0] * (1.0f / kD) - mean * mean;
  float rstd = rsqrtf(var + 1e-5f);
  for (int d = t; d < kD; d += 256) c[d] = (xr[d] - mean) * rstd * pg[d] + pb[d];
  __syncthreads();

  for (int j = t; j < kD; j += 256) {
    float a = b1[j];
    for (int k = 0; k < kD; ++k) a += c[k] * w1[(size_t)k * kD + j];
    tt[j] = tanhf(a);
  }
  __syncthreads();

  for (int o = t; o < kOUT; o += 256) {
    float a = b2[o];
    for (int k = 0; k < kD; ++k) a += tt[k] * w2[(size_t)k * kOUT + o];
    out[(size_t)b * kOUT + o] = a;
  }
}

// ---------------- host orchestration ----------------
extern "C" void kernel_launch(void* const* d_in, const int* in_sizes, int n_in,
                              void* d_out, int out_size, void* d_ws, size_t ws_size,
                              hipStream_t stream) {
  const float* x      = (const float*)d_in[0];
  const float* emb_w  = (const float*)d_in[1];
  const float* emb_b  = (const float*)d_in[2];
  const float* cls    = (const float*)d_in[3];
  const float* pos    = (const float*)d_in[4];
  const float* pre_g  = (const float*)d_in[5];
  const float* pre_b  = (const float*)d_in[6];
  const float* ln1_g  = (const float*)d_in[7];
  const float* ln1_b  = (const float*)d_in[8];
  const float* qkv_w  = (const float*)d_in[9];
  const float* qkv_b  = (const float*)d_in[10];
  const float* proj_w = (const float*)d_in[11];
  const float* proj_b = (const float*)d_in[12];
  const float* ln2_g  = (const float*)d_in[13];
  const float* ln2_b  = (const float*)d_in[14];
  const float* fc1_w  = (const float*)d_in[15];
  const float* fc1_b  = (const float*)d_in[16];
  const float* fc2_w  = (const float*)d_in[17];
  const float* fc2_b  = (const float*)d_in[18];
  const float* post_g = (const float*)d_in[19];
  const float* post_b = (const float*)d_in[20];
  const float* h1w    = (const float*)d_in[21];
  const float* h1b    = (const float*)d_in[22];
  const float* h2w    = (const float*)d_in[23];
  const float* h2b    = (const float*)d_in[24];
  float* out = (float*)d_out;

  // ---- carve workspace ----
  char* cur = (char*)d_ws;
  auto carve = [&](size_t bytes) -> char* {
    char* p = cur;
    cur += (bytes + 255) & ~(size_t)255;
    return p;
  };
  bf16* emb_wt  = (bf16*)carve((size_t)768 * kD * 2);
  bf16* qkv_wt  = (bf16*)carve((size_t)kL * 3 * kD * kD * 2);       // [l][3D][D]
  bf16* proj_wt = (bf16*)carve((size_t)kL * kD * kD * 2);           // [l][D][D]
  bf16* fc1_wt  = (bf16*)carve((size_t)kL * kDFF * kD * 2);         // [l][DFF][D]
  bf16* fc2_wt  = (bf16*)carve((size_t)kL * kD * kDFF * 2);         // [l][D][DFF]
  bf16* patches = (bf16*)carve((size_t)kMP * kD * 2);
  bf16* ybuf    = (bf16*)carve((size_t)kM * kD * 2);
  bf16* ctxbuf  = (bf16*)carve((size_t)kM * kD * 2);
  bf16* mlpbuf  = (bf16*)carve((size_t)kM * kDFF * 2);
  float* tokens = (float*)carve((size_t)kMP * kD * 4);
  float* tmpbuf = (float*)carve((size_t)kM * kD * 4);
  float* hbuf   = (float*)carve((size_t)kM * kD * 4);
  float* qkvbuf = (float*)carve((size_t)kM * 3 * kD * 4);

  auto tr = [&](const float* src, bf16* dst, int K, int N) {
    int total = K * N;
    transpose_bf16_kernel<<<(total + 255) / 256, 256, 0, stream>>>(src, dst, K, N);
  };
  auto gemm = [&](int epi, const bf16* A, const bf16* Bt, const float* bias,
                  float* Cf, bf16* Cb, int M, int N, int K) {
    int tiles = (M >> 5) * (N >> 6);         // 32x64 wave tiles
    int blocks = (tiles + 7) / 8;
    if (epi == 0)      gemm_bf16_kernel<0><<<blocks, 256, 0, stream>>>(A, Bt, bias, Cf, Cb, M, N, K);
    else if (epi == 1) gemm_bf16_kernel<1><<<blocks, 256, 0, stream>>>(A, Bt, bias, Cf, Cb, M, N, K);
    else               gemm_bf16_kernel<2><<<blocks, 256, 0, stream>>>(A, Bt, bias, Cf, Cb, M, N, K);
  };

  // ---- weight conversion (fp32 -> bf16, transposed to [N][K]) ----
  tr(emb_w, emb_wt, 768, kD);
  for (int l = 0; l < kL; ++l) {
    tr(qkv_w  + (size_t)l * kD * 3 * kD, qkv_wt  + (size_t)l * 3 * kD * kD, kD,   3 * kD);
    tr(proj_w + (size_t)l * kD * kD,     proj_wt + (size_t)l * kD * kD,     kD,   kD);
    tr(fc1_w  + (size_t)l * kD * kDFF,   fc1_wt  + (size_t)l * kDFF * kD,   kD,   kDFF);
    tr(fc2_w  + (size_t)l * kDFF * kD,   fc2_wt  + (size_t)l * kD * kDFF,   kDFF, kD);
  }

  // ---- patch embedding ----
  {
    int total = kMP * kD;
    patchify_kernel<<<(total + 255) / 256, 256, 0, stream>>>(x, patches);
  }
  gemm(0, patches, emb_wt, emb_b, tokens, nullptr, kMP, kD, 768);
  {
    int total = kM * kD;
    assemble_kernel<<<(total + 255) / 256, 256, 0, stream>>>(tokens, cls, pos, tmpbuf);
  }
  layernorm_kernel<0><<<kM, 256, 0, stream>>>(tmpbuf, pre_g, pre_b, hbuf, nullptr);

  // ---- transformer blocks ----
  for (int l = 0; l < kL; ++l) {
    layernorm_kernel<1><<<kM, 256, 0, stream>>>(hbuf, ln1_g + l * kD, ln1_b + l * kD, nullptr, ybuf);
    gemm(0, ybuf, qkv_wt + (size_t)l * 3 * kD * kD, qkv_b + (size_t)l * 3 * kD,
         qkvbuf, nullptr, kM, 3 * kD, kD);
    attention_kernel<<<kB * kNH * kS, 64, 0, stream>>>(qkvbuf, ctxbuf);
    gemm(1, ctxbuf, proj_wt + (size_t)l * kD * kD, proj_b + (size_t)l * kD,
         hbuf, nullptr, kM, kD, kD);
    layernorm_kernel<1><<<kM, 256, 0, stream>>>(hbuf, ln2_g + l * kD, ln2_b + l * kD, nullptr, ybuf);
    gemm(2, ybuf, fc1_wt + (size_t)l * kDFF * kD, fc1_b + (size_t)l * kDFF,
         nullptr, mlpbuf, kM, kDFF, kD);
    gemm(1, mlpbuf, fc2_wt + (size_t)l * kD * kDFF, fc2_b + (size_t)l * kD,
         hbuf, nullptr, kM, kD, kDFF);
  }

  // ---- classification head ----
  head_kernel<<<kB, 256, 0, stream>>>(hbuf, post_g, post_b, h1w, h1b, h2w, h2b, out);
}
